// MambaLayer_67851893342294
// MI455X (gfx1250) — compile-verified, matrix operand order checked by reading, not run on hardware
//
#include <hip/hip_runtime.h>
#include <hip/hip_bf16.h>
#include <math.h>

#define D_MODEL 1024
#define D_STATE 16
#define D_CONV  4
#define D_INNER 2048
#define DT_RANK 64
#define B_SZ    2
#define L_SEQ   2048
#define NROWS   (B_SZ * L_SEQ)        // 4096
#define XDB_W   (DT_RANK + 2*D_STATE) // 96
#define NCHUNK  16
#define CLEN    (L_SEQ / NCHUNK)      // 128

typedef __attribute__((ext_vector_type(16))) __bf16 bf16x16;
typedef __attribute__((ext_vector_type(8)))  __bf16 bf16x8;
typedef __attribute__((ext_vector_type(8)))  float  f32x8;

#define BM 128
#define BN 128
#define BK 32
#define LDSS 40   // BK + 8 pad (80B row stride, 16B aligned)

enum { EPI_STORE = 0, EPI_BIAS_SOFTPLUS = 1, EPI_ACCUM = 2 };

__device__ __forceinline__ float sigmoidf_(float v) { return 1.f / (1.f + __expf(-v)); }

// ---------------------------------------------------------------------------
// Generic GEMM:  C[M,N] = epi( A[M,K] * W[N,K]^T )   (A row-major, W row-major
// [N,K] as the reference weights are stored).  bf16 WMMA with f32 accumulate.
// M must be a multiple of 128, K a multiple of 32; N is guarded.
// ---------------------------------------------------------------------------
__global__ __launch_bounds__(256) void gemm_wmma_bf16(
    const float* __restrict__ A, const float* __restrict__ W,
    float* __restrict__ C, const float* __restrict__ bias,
    int M, int N, int K, int lda, int ldw, int ldc, int epi)
{
  __shared__ __bf16 As[BM][LDSS];
  __shared__ __bf16 Ws[BN][LDSS];

  const int tid  = threadIdx.x;
  const int lane = tid & 31;
  const int wid  = tid >> 5;        // 8 waves
  const int wm   = wid & 3;         // 4 waves along M (32 rows each)
  const int wn   = wid >> 2;        // 2 waves along N (64 cols each)
  const int half = lane >> 4;       // K-group select per ISA bf16 layout
  const int l16  = lane & 15;

  const int mBlock = blockIdx.y * BM;
  const int nBlock = blockIdx.x * BN;

  f32x8 acc[2][4];
#pragma unroll
  for (int mi = 0; mi < 2; ++mi)
#pragma unroll
    for (int ni = 0; ni < 4; ++ni)
      acc[mi][ni] = {};

  const int stg_r   = tid >> 1;     // 0..127 : tile row
  const int stg_seg = tid & 1;      // 0..1   : 16-wide K segment

  const int KT = K / BK;
  for (int kt = 0; kt < KT; ++kt) {
    const int k0 = kt * BK;
    __syncthreads();

    // ---- stage A tile (f32 -> bf16) ----
    {
      const float* ap = A + (size_t)(mBlock + stg_r) * lda + (k0 + stg_seg * 16);
      float4 f0 = ((const float4*)ap)[0];
      float4 f1 = ((const float4*)ap)[1];
      float4 f2 = ((const float4*)ap)[2];
      float4 f3 = ((const float4*)ap)[3];
      if (k0 + BK < K) __builtin_prefetch(ap + BK, 0, 3);  // near-cache prefetch
      union { bf16x8 v; __bf16 e[8]; } p0, p1;
      p0.e[0] = (__bf16)f0.x; p0.e[1] = (__bf16)f0.y; p0.e[2] = (__bf16)f0.z; p0.e[3] = (__bf16)f0.w;
      p0.e[4] = (__bf16)f1.x; p0.e[5] = (__bf16)f1.y; p0.e[6] = (__bf16)f1.z; p0.e[7] = (__bf16)f1.w;
      p1.e[0] = (__bf16)f2.x; p1.e[1] = (__bf16)f2.y; p1.e[2] = (__bf16)f2.z; p1.e[3] = (__bf16)f2.w;
      p1.e[4] = (__bf16)f3.x; p1.e[5] = (__bf16)f3.y; p1.e[6] = (__bf16)f3.z; p1.e[7] = (__bf16)f3.w;
      *(bf16x8*)&As[stg_r][stg_seg * 16]     = p0.v;
      *(bf16x8*)&As[stg_r][stg_seg * 16 + 8] = p1.v;
    }
    // ---- stage W tile (f32 -> bf16), zero-fill rows >= N ----
    {
      const int n = nBlock + stg_r;
      float4 f0 = {0.f, 0.f, 0.f, 0.f}, f1 = f0, f2 = f0, f3 = f0;
      if (n < N) {
        const float* wp = W + (size_t)n * ldw + (k0 + stg_seg * 16);
        f0 = ((const float4*)wp)[0];
        f1 = ((const float4*)wp)[1];
        f2 = ((const float4*)wp)[2];
        f3 = ((const float4*)wp)[3];
        if (k0 + BK < K) __builtin_prefetch(wp + BK, 0, 3);
      }
      union { bf16x8 v; __bf16 e[8]; } p0, p1;
      p0.e[0] = (__bf16)f0.x; p0.e[1] = (__bf16)f0.y; p0.e[2] = (__bf16)f0.z; p0.e[3] = (__bf16)f0.w;
      p0.e[4] = (__bf16)f1.x; p0.e[5] = (__bf16)f1.y; p0.e[6] = (__bf16)f1.z; p0.e[7] = (__bf16)f1.w;
      p1.e[0] = (__bf16)f2.x; p1.e[1] = (__bf16)f2.y; p1.e[2] = (__bf16)f2.z; p1.e[3] = (__bf16)f2.w;
      p1.e[4] = (__bf16)f3.x; p1.e[5] = (__bf16)f3.y; p1.e[6] = (__bf16)f3.z; p1.e[7] = (__bf16)f3.w;
      *(bf16x8*)&Ws[stg_r][stg_seg * 16]     = p0.v;
      *(bf16x8*)&Ws[stg_r][stg_seg * 16 + 8] = p1.v;
    }
    __syncthreads();

    // ---- build fragments per ISA 16-bit layout and issue WMMA ----
    bf16x16 af[2], bfg[4];
#pragma unroll
    for (int mi = 0; mi < 2; ++mi) {
      const int r = wm * 32 + mi * 16 + l16;
      bf16x8 lo = *(const bf16x8*)&As[r][half * 8];
      bf16x8 hi = *(const bf16x8*)&As[r][16 + half * 8];
      af[mi] = __builtin_shufflevector(lo, hi, 0,1,2,3,4,5,6,7,8,9,10,11,12,13,14,15);
    }
#pragma unroll
    for (int ni = 0; ni < 4; ++ni) {
      const int r = wn * 64 + ni * 16 + l16;
      bf16x8 lo = *(const bf16x8*)&Ws[r][half * 8];
      bf16x8 hi = *(const bf16x8*)&Ws[r][16 + half * 8];
      bfg[ni] = __builtin_shufflevector(lo, hi, 0,1,2,3,4,5,6,7,8,9,10,11,12,13,14,15);
    }
#pragma unroll
    for (int mi = 0; mi < 2; ++mi)
#pragma unroll
      for (int ni = 0; ni < 4; ++ni)
        acc[mi][ni] = __builtin_amdgcn_wmma_f32_16x16x32_bf16(
            false, af[mi], false, bfg[ni], (short)0, acc[mi][ni], false, false);
  }

  // ---- epilogue (C layout: row = m0 + half*8 + v, col = n0 + l16) ----
#pragma unroll
  for (int mi = 0; mi < 2; ++mi) {
#pragma unroll
    for (int ni = 0; ni < 4; ++ni) {
      const int m0 = mBlock + wm * 32 + mi * 16;
      const int n0 = nBlock + wn * 64 + ni * 16;
      const int gc = n0 + l16;
      union { f32x8 v; float e[8]; } u;
      u.v = acc[mi][ni];
      if (gc < N) {
#pragma unroll
        for (int v = 0; v < 8; ++v) {
          const int gr = m0 + half * 8 + v;
          float val = u.e[v];
          const size_t off = (size_t)gr * ldc + gc;
          if (epi == EPI_BIAS_SOFTPLUS) {
            val += bias[gc];
            C[off] = (val > 20.f) ? val : log1pf(__expf(val));
          } else if (epi == EPI_ACCUM) {
            C[off] += val;
          } else {
            C[off] = val;
          }
        }
      }
    }
  }
}

// ---------------------------------------------------------------------------
// Depthwise causal conv1d (+bias, +SiLU).  rev=1 runs the time-reversed block
// on the original axis: reversed kernel, anti-causal window.
// ---------------------------------------------------------------------------
__global__ __launch_bounds__(256) void conv_silu_kernel(
    const float* __restrict__ xz, const float* __restrict__ conv_w,
    const float* __restrict__ conv_b, float* __restrict__ xc, int rev)
{
  const size_t idx = (size_t)blockIdx.x * blockDim.x + threadIdx.x;
  if (idx >= (size_t)NROWS * D_INNER) return;
  const int d   = (int)(idx % D_INNER);
  const int row = (int)(idx / D_INNER);
  const int t   = row % L_SEQ;
  const int b   = row / L_SEQ;
  float acc = conv_b[d];
#pragma unroll
  for (int j = 0; j < D_CONV; ++j) {
    const int st = rev ? (t + (D_CONV - 1) - j) : (t - (D_CONV - 1) + j);
    if (st >= 0 && st < L_SEQ)
      acc += conv_w[d * D_CONV + j] *
             xz[(size_t)(b * L_SEQ + st) * (2 * D_INNER) + d];
  }
  xc[(size_t)row * D_INNER + d] = acc * sigmoidf_(acc);
}

// ---------------------------------------------------------------------------
// Chunked parallel scan for  h_t = a_t * h_{t-1} + b_t,
//   a_t = exp(delta_t * A),  b_t = (delta_t * u_t) * B_t.
// Scan position p (0..L-1) maps to time t = rev ? L-1-p : p.
// Pass 1: per-chunk summaries  A_c = prod(a),  B_c = fold of b.
// ---------------------------------------------------------------------------
__global__ __launch_bounds__(256) void scan_pass1(
    const float* __restrict__ delta, const float* __restrict__ xdb,
    const float* __restrict__ xc,    const float* __restrict__ A_log,
    float* __restrict__ Achunk, float* __restrict__ Bchunk, int rev)
{
  const int d = blockIdx.x * blockDim.x + threadIdx.x;
  const int c = blockIdx.y;
  const int b = blockIdx.z;
  if (d >= D_INNER) return;

  float Ac[D_STATE];
#pragma unroll
  for (int s = 0; s < D_STATE; ++s) Ac[s] = -__expf(A_log[(size_t)d * D_STATE + s]);

  float ap[D_STATE], bp[D_STATE];
#pragma unroll
  for (int s = 0; s < D_STATE; ++s) { ap[s] = 1.f; bp[s] = 0.f; }

  for (int i = 0; i < CLEN; ++i) {
    const int p   = c * CLEN + i;
    const int t   = rev ? (L_SEQ - 1 - p) : p;
    const int row = b * L_SEQ + t;
    const float dlt = delta[(size_t)row * D_INNER + d];
    const float u   = xc[(size_t)row * D_INNER + d];
    const float du  = dlt * u;
    const float* bc = xdb + (size_t)row * XDB_W + DT_RANK;
#pragma unroll
    for (int s = 0; s < D_STATE; ++s) {
      const float a = __expf(dlt * Ac[s]);
      bp[s] = a * bp[s] + du * bc[s];
      ap[s] *= a;
    }
  }
  const size_t o = (((size_t)b * NCHUNK + c) * D_INNER + d) * D_STATE;
#pragma unroll
  for (int s = 0; s < D_STATE; ++s) { Achunk[o + s] = ap[s]; Bchunk[o + s] = bp[s]; }
}

// Pass 2: serial carry over the 16 chunks, parallel over all (b,d,s).
__global__ __launch_bounds__(256) void scan_pass2(
    const float* __restrict__ Achunk, const float* __restrict__ Bchunk,
    float* __restrict__ hin)
{
  const int idx = blockIdx.x * blockDim.x + threadIdx.x;   // 0 .. B*D_INNER*D_STATE-1
  const int b = idx / (D_INNER * D_STATE);
  const int r = idx % (D_INNER * D_STATE);
  float h = 0.f;
#pragma unroll
  for (int c = 0; c < NCHUNK; ++c) {
    const size_t o = ((size_t)b * NCHUNK + c) * ((size_t)D_INNER * D_STATE) + r;
    hin[o] = h;
    h = Achunk[o] * h + Bchunk[o];
  }
}

// Pass 3: rerun each chunk from its incoming state; emit gated output.
__global__ __launch_bounds__(256) void scan_pass3(
    const float* __restrict__ delta, const float* __restrict__ xdb,
    const float* __restrict__ xc,    const float* __restrict__ xz,
    const float* __restrict__ A_log, const float* __restrict__ Dp,
    const float* __restrict__ hin,   float* __restrict__ ybuf, int rev)
{
  const int d = blockIdx.x * blockDim.x + threadIdx.x;
  const int c = blockIdx.y;
  const int b = blockIdx.z;
  if (d >= D_INNER) return;

  float Ac[D_STATE];
#pragma unroll
  for (int s = 0; s < D_STATE; ++s) Ac[s] = -__expf(A_log[(size_t)d * D_STATE + s]);

  float h[D_STATE];
  const size_t o = (((size_t)b * NCHUNK + c) * D_INNER + d) * D_STATE;
#pragma unroll
  for (int s = 0; s < D_STATE; ++s) h[s] = hin[o + s];
  const float Dd = Dp[d];

  for (int i = 0; i < CLEN; ++i) {
    const int p   = c * CLEN + i;
    const int t   = rev ? (L_SEQ - 1 - p) : p;
    const int row = b * L_SEQ + t;
    const float dlt = delta[(size_t)row * D_INNER + d];
    const float u   = xc[(size_t)row * D_INNER + d];
    const float z   = xz[(size_t)row * (2 * D_INNER) + D_INNER + d];
    const float du  = dlt * u;
    const float* bc = xdb + (size_t)row * XDB_W + DT_RANK;
    float y = 0.f;
#pragma unroll
    for (int s = 0; s < D_STATE; ++s) {
      h[s] = __expf(dlt * Ac[s]) * h[s] + du * bc[s];
      y += h[s] * bc[D_STATE + s];
    }
    ybuf[(size_t)row * D_INNER + d] = (y + u * Dd) * (z * sigmoidf_(z));
  }
}

// ---------------------------------------------------------------------------
extern "C" void kernel_launch(void* const* d_in, const int* in_sizes, int n_in,
                              void* d_out, int out_size, void* d_ws, size_t ws_size,
                              hipStream_t stream) {
  (void)in_sizes; (void)n_in; (void)out_size; (void)ws_size;
  const float* x = (const float*)d_in[0];

  const size_t CH = (size_t)B_SZ * NCHUNK * D_INNER * D_STATE;   // 1 Mi floats
  float* ws  = (float*)d_ws;
  float* xz  = ws;                                    // [4096, 4096]
  float* xc  = xz  + (size_t)NROWS * (2 * D_INNER);   // [4096, 2048]
  float* xdb = xc  + (size_t)NROWS * D_INNER;         // [4096, 96]
  float* dl  = xdb + (size_t)NROWS * XDB_W;           // [4096, 2048]
  float* yb  = dl  + (size_t)NROWS * D_INNER;         // [4096, 2048]
  float* Achk = yb   + (size_t)NROWS * D_INNER;       // [2,16,2048,16]
  float* Bchk = Achk + CH;
  float* hin  = Bchk + CH;
  float* out = (float*)d_out;                         // [4096, 1024]

  const dim3 blk(256);
  const dim3 scan_grid(D_INNER / 256, NCHUNK, B_SZ);
  for (int dir = 0; dir < 2; ++dir) {
    const int base = 1 + dir * 9;
    const float* in_w    = (const float*)d_in[base + 0];
    const float* conv_w  = (const float*)d_in[base + 1];
    const float* conv_b  = (const float*)d_in[base + 2];
    const float* xproj_w = (const float*)d_in[base + 3];
    const float* dt_w    = (const float*)d_in[base + 4];
    const float* dt_b    = (const float*)d_in[base + 5];
    const float* A_log   = (const float*)d_in[base + 6];
    const float* Dp      = (const float*)d_in[base + 7];
    const float* out_w   = (const float*)d_in[base + 8];
    const int rev = dir;

    // 1) in_proj: xz = x @ in_w^T            [4096,1024]x[4096,1024]^T
    gemm_wmma_bf16<<<dim3((2 * D_INNER + BN - 1) / BN, NROWS / BM), blk, 0, stream>>>(
        x, in_w, xz, nullptr, NROWS, 2 * D_INNER, D_MODEL, D_MODEL, D_MODEL,
        2 * D_INNER, EPI_STORE);

    // 2) depthwise conv + bias + SiLU
    conv_silu_kernel<<<dim3((unsigned)(((size_t)NROWS * D_INNER) / 256)), blk, 0, stream>>>(
        xz, conv_w, conv_b, xc, rev);

    // 3) x_proj: xdb = xc @ xproj_w^T        N=96 (guarded)
    gemm_wmma_bf16<<<dim3((XDB_W + BN - 1) / BN, NROWS / BM), blk, 0, stream>>>(
        xc, xproj_w, xdb, nullptr, NROWS, XDB_W, D_INNER, D_INNER, D_INNER,
        XDB_W, EPI_STORE);

    // 4) dt_proj + softplus: dl = softplus(xdb[:, :64] @ dt_w^T + dt_b)
    gemm_wmma_bf16<<<dim3((D_INNER + BN - 1) / BN, NROWS / BM), blk, 0, stream>>>(
        xdb, dt_w, dl, dt_b, NROWS, D_INNER, DT_RANK, XDB_W, DT_RANK,
        D_INNER, EPI_BIAS_SOFTPLUS);

    // 5) chunked parallel scan (3 passes) + skip + gate
    scan_pass1<<<scan_grid, blk, 0, stream>>>(dl, xdb, xc, A_log, Achk, Bchk, rev);
    scan_pass2<<<dim3((unsigned)(CH / (NCHUNK * 256))), blk, 0, stream>>>(Achk, Bchk, hin);
    scan_pass3<<<scan_grid, blk, 0, stream>>>(dl, xdb, xc, xz, A_log, Dp, hin, yb, rev);

    // 6) out_proj: out (+)= yb @ out_w^T
    gemm_wmma_bf16<<<dim3((D_MODEL + BN - 1) / BN, NROWS / BM), blk, 0, stream>>>(
        yb, out_w, out, nullptr, NROWS, D_MODEL, D_INNER, D_INNER, D_INNER,
        D_MODEL, dir == 0 ? EPI_STORE : EPI_ACCUM);
  }
}
